// CrystalGATVAE_84129819394657
// MI455X (gfx1250) — matrix operand order repaired and checked by reading, not compiled, not run on hardware
//
#include <hip/hip_runtime.h>
#include <hip/hip_bf16.h>
#include <math.h>

// ---------------------------------------------------------------------------
// CrystalGATVAE forward for MI455X (gfx1250, wave32, WMMA).
//
// Algebraic optimizations:
//  * temp[n,:fin] = proj[n] * degree(n)      (kills the big scatter-add)
//  * edge decoder ed1: only B=8 distinct latent rows -> precompute
//      ztop = zp@W_top, zbot = zp@W_bot and fuse per-edge   (21 GFLOP -> ~0.1)
//  * node decoder nd1: (z_exp + nep)@W = nep@W + (zp@W)[graph_id]
//  * All K>=32 GEMMs run a f16-fragment / f32-accumulate WMMA kernel:
//      128x64 block tile, 8 waves, 2x2 16x16 WMMA tiles per wave.
//      A staged row-major, B staged TRANSPOSED in LDS so every fragment is
//      two aligned ds_load_b128; interior tiles use vectorized global loads.
// ---------------------------------------------------------------------------

typedef __attribute__((ext_vector_type(16))) _Float16 v16h;
typedef __attribute__((ext_vector_type(8)))  _Float16 h8;
typedef __attribute__((ext_vector_type(8)))  float    v8f;

#define WBM 128
#define WBN 64
#define WBK 32
#define A_STRIDE 40   // halfs per LDS A row (80B: multiple of 16B)
#define BT_STRIDE 40  // halfs per LDS Bt row (col-major over k)

union V16 { v16h v; h8 h[2]; };

// --------------------------- WMMA GEMM kernel ------------------------------
// C[M,Nc] = A[M,K](lda) @ B[K,Nc](ldb) (+bias[j]) (optional ReLU).
__global__ __launch_bounds__(256) void gemm_wmma_kernel(
    const float* __restrict__ A, int lda,
    const float* __restrict__ B, int ldb,
    const float* __restrict__ bias, float* __restrict__ C,
    int M, int K, int Nc, int relu)
{
    __shared__ _Float16 As[WBM * A_STRIDE];
    __shared__ _Float16 Bt[WBN * BT_STRIDE];   // transposed: Bt[col][k]

    const int tid  = threadIdx.x;
    const int lane = tid & 31;
    const int wave = tid >> 5;
    const int wm   = wave >> 1;   // 0..3 (M direction, 32 rows each)
    const int wn   = wave & 1;    // 0..1 (N direction, 32 cols each)
    const int blockM = blockIdx.y * WBM;
    const int blockN = blockIdx.x * WBN;

    const bool fullM = (blockM + WBM <= M);
    const bool fullN = (blockN + WBN <= Nc);

    v8f acc00 = {}, acc01 = {}, acc10 = {}, acc11 = {};

    for (int k0 = 0; k0 < K; k0 += WBK) {
        const bool fullK = (k0 + WBK <= K);

        // ---- stage A (128x32, row-major) ----
        if (fullM && fullK) {
            #pragma unroll
            for (int t = 0; t < 4; ++t) {
                int i = tid + t * 256;
                int r = i >> 3, cv = (i & 7) << 2;
                const float4 v = *(const float4*)(A + (long)(blockM + r) * lda + (k0 + cv));
                union { _Float16 h[4]; unsigned long long u; } pk;
                pk.h[0] = (_Float16)v.x; pk.h[1] = (_Float16)v.y;
                pk.h[2] = (_Float16)v.z; pk.h[3] = (_Float16)v.w;
                *(unsigned long long*)&As[r * A_STRIDE + cv] = pk.u;
            }
        } else {
            for (int i = tid; i < WBM * WBK; i += 256) {
                int r = i >> 5, c = i & 31;
                int gr = blockM + r, gc = k0 + c;
                float v = (gr < M && gc < K) ? A[(long)gr * lda + gc] : 0.0f;
                As[r * A_STRIDE + c] = (_Float16)v;
            }
        }
        // ---- stage B (32x64) transposed into Bt[col][k] ----
        if (fullN && fullK) {
            // 256 threads = 64 cols x 4 k-chunks of 8: coalesced b32 loads,
            // one ds_store_b128 per thread.
            int col = tid & 63;
            int kc  = (tid >> 6) << 3;
            union { _Float16 h[8]; h8 v; } pk;
            #pragma unroll
            for (int i = 0; i < 8; ++i)
                pk.h[i] = (_Float16)B[(long)(k0 + kc + i) * ldb + (blockN + col)];
            *(h8*)&Bt[col * BT_STRIDE + kc] = pk.v;
        } else {
            for (int i = tid; i < WBK * WBN; i += 256) {
                int c = i & 63, r = i >> 6;
                int gr = k0 + r, gc = blockN + c;
                float v = (gr < K && gc < Nc) ? B[(long)gr * ldb + gc] : 0.0f;
                Bt[c * BT_STRIDE + r] = (_Float16)v;
            }
        }
        __syncthreads();

        // ---- fragments per ISA 7.12.2: all gathers are aligned b128 loads ----
        // A frag: a[0..7] = As[row][kbA..kbA+7], a[8..15] = As[row][kbA+16..+23]
        // B frag: b[0..7] = Bt[col][kbB..kbB+7], b[8..15] = Bt[col][kbB+16..+23]
        const int rowA = lane & 15;
        const int kbA  = (lane < 16) ? 0 : 8;
        const int colB = wn * 32 + (lane & 15);
        const int kbB  = (lane < 16) ? 0 : 8;
        const _Float16* ap0 = &As[(wm * 32 + rowA) * A_STRIDE + kbA];
        const _Float16* ap1 = &As[(wm * 32 + 16 + rowA) * A_STRIDE + kbA];
        const _Float16* bp0 = &Bt[colB * BT_STRIDE + kbB];
        const _Float16* bp1 = &Bt[(colB + 16) * BT_STRIDE + kbB];
        V16 a0, a1, b0, b1;
        a0.h[0] = *(const h8*)ap0;        a0.h[1] = *(const h8*)(ap0 + 16);
        a1.h[0] = *(const h8*)ap1;        a1.h[1] = *(const h8*)(ap1 + 16);
        b0.h[0] = *(const h8*)bp0;        b0.h[1] = *(const h8*)(bp0 + 16);
        b1.h[0] = *(const h8*)bp1;        b1.h[1] = *(const h8*)(bp1 + 16);

        // 4 independent WMMAs: A frags reused across N, B frags across M
        acc00 = __builtin_amdgcn_wmma_f32_16x16x32_f16(false, a0.v, false, b0.v, (short)0, acc00, false, false);
        acc01 = __builtin_amdgcn_wmma_f32_16x16x32_f16(false, a0.v, false, b1.v, (short)0, acc01, false, false);
        acc10 = __builtin_amdgcn_wmma_f32_16x16x32_f16(false, a1.v, false, b0.v, (short)0, acc10, false, false);
        acc11 = __builtin_amdgcn_wmma_f32_16x16x32_f16(false, a1.v, false, b1.v, (short)0, acc11, false, false);
        __syncthreads();
    }

    // ---- store 4 tiles (C/D layout: VGPR r -> M = r / r+8 across lane halves) ----
    const int nc0 = lane & 15;
    const int mB  = (lane < 16) ? 0 : 8;
    auto store_tile = [&](v8f accv, int moff, int noff) {
        int gc = blockN + wn * 32 + noff + nc0;
        if (gc >= Nc) return;
        float bv = bias ? bias[gc] : 0.0f;
        #pragma unroll
        for (int r = 0; r < 8; ++r) {
            int gr = blockM + wm * 32 + moff + mB + r;
            if (gr < M) {
                float v = accv[r] + bv;
                if (relu) v = fmaxf(v, 0.0f);
                C[(long)gr * Nc + gc] = v;
            }
        }
    };
    store_tile(acc00, 0, 0);
    store_tile(acc01, 0, 16);
    store_tile(acc10, 16, 0);
    store_tile(acc11, 16, 16);
}

// --------------------------- helper device ops -----------------------------

__device__ __forceinline__ void atomicMaxFloat(float* addr, float v) {
    if (v >= 0.0f) atomicMax((int*)addr, __float_as_int(v));
    else           atomicMin((unsigned int*)addr, __float_as_uint(v));
}

__global__ void fill_kernel(float* p, long n, float v) {
    long i = (long)blockIdx.x * blockDim.x + threadIdx.x;
    if (i < n) p[i] = v;
}

__global__ void small_linear_kernel(const float* __restrict__ A,
                                    const float* __restrict__ W,
                                    const float* __restrict__ bias,
                                    float* __restrict__ C,
                                    int M, int K, int Nc, int relu)
{
    long i = (long)blockIdx.x * blockDim.x + threadIdx.x;
    if (i >= (long)M * Nc) return;
    int m = (int)(i / Nc), j = (int)(i % Nc);
    float acc = bias ? bias[j] : 0.0f;
    const float* a = A + (long)m * K;
    for (int k = 0; k < K; ++k) acc += a[k] * W[(long)k * Nc + j];
    if (relu) acc = fmaxf(acc, 0.0f);
    C[i] = acc;
}

__global__ void edge_mlp_kernel(const float* __restrict__ ef_raw,
                                const float* __restrict__ w1, const float* __restrict__ b1,
                                const float* __restrict__ w2, const float* __restrict__ b2,
                                float* __restrict__ ef, int E)
{
    int e = blockIdx.x * blockDim.x + threadIdx.x;
    if (e >= E) return;
    float x0 = ef_raw[e * 3], x1 = ef_raw[e * 3 + 1], x2 = ef_raw[e * 3 + 2];
    float hbuf[6];
    #pragma unroll
    for (int j = 0; j < 6; ++j) {
        float v = b1[j] + x0 * w1[j] + x1 * w1[6 + j] + x2 * w1[12 + j];
        hbuf[j] = fmaxf(v, 0.0f);
    }
    #pragma unroll
    for (int j = 0; j < 3; ++j) {
        float v = b2[j];
        #pragma unroll
        for (int k = 0; k < 6; ++k) v += hbuf[k] * w2[k * 3 + j];
        ef[e * 3 + j] = v;
    }
}

__global__ void deg_kernel(const int* __restrict__ src, const int* __restrict__ dst,
                           float* __restrict__ deg, int E)
{
    int e = blockIdx.x * blockDim.x + threadIdx.x;
    if (e >= E) return;
    atomicAdd(&deg[src[e]], 1.0f);
    atomicAdd(&deg[dst[e]], 1.0f);
}

__global__ void temp_proj_kernel(const float* __restrict__ proj, const float* __restrict__ deg,
                                 float* __restrict__ temp, int N, int fin, int ldt)
{
    long i = (long)blockIdx.x * blockDim.x + threadIdx.x;
    if (i >= (long)N * fin) return;
    int n = (int)(i / fin), j = (int)(i % fin);
    temp[(long)n * ldt + j] = proj[i] * deg[n];
}

__global__ void temp_edge_kernel(const float* __restrict__ ef, const int* __restrict__ src,
                                 const int* __restrict__ dst, float* __restrict__ temp,
                                 int E, int fin, int ldt)
{
    int e = blockIdx.x * blockDim.x + threadIdx.x;
    if (e >= E) return;
    int s = src[e], d = dst[e];
    #pragma unroll
    for (int j = 0; j < 3; ++j) {
        float v = ef[e * 3 + j];
        atomicAdd(&temp[(long)s * ldt + fin + j], v);
        atomicAdd(&temp[(long)d * ldt + fin + j], v);
    }
}

__global__ void attn_lr_kernel(const float* __restrict__ hpre, const float* __restrict__ al,
                               const float* __restrict__ ar, float* __restrict__ el,
                               float* __restrict__ er, int N, int outd)
{
    int i = blockIdx.x * blockDim.x + threadIdx.x;
    if (i >= N * 4) return;
    int n = i >> 2, hd = i & 3;
    const float* hp  = hpre + ((long)n * 4 + hd) * outd;
    const float* alp = al + hd * outd;
    const float* arp = ar + hd * outd;
    float sl = 0.0f, sr = 0.0f;
    for (int o = 0; o < outd; ++o) { float hv = hp[o]; sl += hv * alp[o]; sr += hv * arp[o]; }
    el[i] = sl; er[i] = sr;
}

__global__ void edge_logit_max_kernel(const int* __restrict__ src, const int* __restrict__ dst,
                                      const float* __restrict__ el, const float* __restrict__ er,
                                      float* __restrict__ elr, float* __restrict__ m, int E)
{
    int e = blockIdx.x * blockDim.x + threadIdx.x;
    if (e >= E) return;
    int s = src[e], d = dst[e];
    #pragma unroll
    for (int hd = 0; hd < 4; ++hd) {
        float v = el[s * 4 + hd] + er[d * 4 + hd];
        v = v > 0.0f ? v : 0.2f * v;                 // leaky_relu(0.2)
        elr[e * 4 + hd] = v;
        atomicMaxFloat(&m[d * 4 + hd], v);
    }
}

__global__ void edge_exp_sum_kernel(const int* __restrict__ dst, float* __restrict__ elr,
                                    const float* __restrict__ m, float* __restrict__ sbuf, int E)
{
    int e = blockIdx.x * blockDim.x + threadIdx.x;
    if (e >= E) return;
    int d = dst[e];
    #pragma unroll
    for (int hd = 0; hd < 4; ++hd) {
        float ex = expf(elr[e * 4 + hd] - m[d * 4 + hd]);
        elr[e * 4 + hd] = ex;
        atomicAdd(&sbuf[d * 4 + hd], ex);
    }
}

__global__ void agg_kernel(const float* __restrict__ hpre, const float* __restrict__ ex,
                           const float* __restrict__ sbuf, const int* __restrict__ src,
                           const int* __restrict__ dst, float* __restrict__ agg,
                           int E, int outd)
{
    long i = (long)blockIdx.x * blockDim.x + threadIdx.x;
    if (i >= (long)E * 4) return;
    int e = (int)(i >> 2), hd = (int)(i & 3);
    int s = src[e], d = dst[e];
    float w = ex[i] / sbuf[d * 4 + hd];
    const float* hp = hpre + ((long)s * 4 + hd) * outd;
    float* ap = agg + ((long)d * 4 + hd) * outd;
    for (int o = 0; o < outd; ++o) atomicAdd(&ap[o], hp[o] * w);
}

__global__ void elu_bias_kernel(float* __restrict__ x, const float* __restrict__ b,
                                long n, int D)
{
    long i = (long)blockIdx.x * blockDim.x + threadIdx.x;
    if (i >= n) return;
    float v = x[i] + b[i % D];
    x[i] = v > 0.0f ? v : expm1f(v);
}

__global__ void graph_max_kernel(const float* __restrict__ gate, const int* __restrict__ gid,
                                 float* __restrict__ gm, int N)
{
    int n = blockIdx.x * blockDim.x + threadIdx.x;
    if (n >= N) return;
    atomicMaxFloat(&gm[gid[n]], gate[n]);
}

__global__ void graph_expsum_kernel(float* __restrict__ gate, const int* __restrict__ gid,
                                    const float* __restrict__ gm, float* __restrict__ gsum, int N)
{
    int n = blockIdx.x * blockDim.x + threadIdx.x;
    if (n >= N) return;
    float v = expf(gate[n] - gm[gid[n]]);
    gate[n] = v;
    atomicAdd(&gsum[gid[n]], v);
}

__global__ void pool_kernel(const float* __restrict__ h, const float* __restrict__ gate,
                            const float* __restrict__ gsum, const int* __restrict__ gid,
                            float* __restrict__ gemb, int N, int D)
{
    long i = (long)blockIdx.x * blockDim.x + threadIdx.x;
    if (i >= (long)N * D) return;
    int n = (int)(i / D), j = (int)(i % D);
    int g = gid[n];
    atomicAdd(&gemb[(long)g * D + j], h[i] * (gate[n] / gsum[g]));
}

__global__ void z_kernel(const float* __restrict__ mu, float* __restrict__ logvar,
                         const float* __restrict__ eps, float* __restrict__ z, int n)
{
    int i = blockIdx.x * blockDim.x + threadIdx.x;
    if (i >= n) return;
    float lv = fminf(fmaxf(logvar[i], -10.0f), 10.0f);
    logvar[i] = lv;
    float stdv = fminf(fmaxf(expf(0.5f * lv), 1e-6f), 1e6f);
    z[i] = mu[i] + eps[i] * stdv;
}

__global__ void add_relu_gather_kernel(float* __restrict__ y, const float* __restrict__ zc,
                                       const int* __restrict__ gid, int N, int D)
{
    long i = (long)blockIdx.x * blockDim.x + threadIdx.x;
    if (i >= (long)N * D) return;
    int n = (int)(i / D), j = (int)(i % D);
    float v = y[i] + zc[(long)gid[n] * D + j];
    y[i] = fmaxf(v, 0.0f);
}

__global__ void edge_decode_kernel(const int* __restrict__ src, const int* __restrict__ dst,
                                   const int* __restrict__ gid, const float* __restrict__ ztop,
                                   const float* __restrict__ zbot, const float* __restrict__ w2,
                                   const float* __restrict__ b2, float* __restrict__ out, int E)
{
    int e = blockIdx.x * blockDim.x + threadIdx.x;
    if (e >= E) return;
    int gs = gid[src[e]], gd = gid[dst[e]];
    const float* t  = ztop + (long)gs * 128;
    const float* bo = zbot + (long)gd * 128;
    float o0 = b2[0], o1 = b2[1], o2 = b2[2];
    for (int k = 0; k < 128; ++k) {
        float v = fmaxf(t[k] + bo[k], 0.0f);
        o0 += v * w2[k * 3];
        o1 += v * w2[k * 3 + 1];
        o2 += v * w2[k * 3 + 2];
    }
    out[e * 3] = o0; out[e * 3 + 1] = o1; out[e * 3 + 2] = o2;
}

// ------------------------------- host side ---------------------------------

static inline unsigned int cdivl(long a, int b) { return (unsigned int)((a + b - 1) / b); }

extern "C" void kernel_launch(void* const* d_in, const int* in_sizes, int n_in,
                              void* d_out, int out_size, void* d_ws, size_t ws_size,
                              hipStream_t stream)
{
    constexpr int NODES = 10000, EDGES = 160000, BG = 8, HH = 4, HID = 128, LATD = 64;

    // --- input indexing: JAX pytree flatten = dict keys sorted at every level ---
    // top: dst, edge_feat, eps, graph_id, node_feat, params(...), src
    // _lin -> {b, w}; _gat -> {al, ar, em1{b,w}, em2{b,w}, gb, gw, np_{b,w}}
    auto F = [&](int i) { return (const float*)d_in[i]; };
    const int*   dstI      = (const int*)d_in[0];
    const float* edge_feat = F(1);
    const float* eps       = F(2);
    const int*   gid       = (const int*)d_in[3];
    const float* node_feat = F(4);
    const float *ed1_b = F(5),  *ed1_w = F(6);
    const float *ed2_b = F(7),  *ed2_w = F(8);
    const float *en1_b = F(9),  *en1_w = F(10);
    const float *en2_b = F(11), *en2_w = F(12);
    const float *flv_b = F(13), *flv_w = F(14);
    const float *fmu_b = F(15), *fmu_w = F(16);
    const float *g1_al = F(17), *g1_ar = F(18), *g1_em1b = F(19), *g1_em1w = F(20),
                *g1_em2b = F(21), *g1_em2w = F(22), *g1_gb = F(23), *g1_gw = F(24),
                *g1_npb = F(25), *g1_npw = F(26);
    const float *g2_al = F(27), *g2_ar = F(28), *g2_em1b = F(29), *g2_em1w = F(30),
                *g2_em2b = F(31), *g2_em2w = F(32), *g2_gb = F(33), *g2_gw = F(34),
                *g2_npb = F(35), *g2_npw = F(36);
    const float *g3_al = F(37), *g3_ar = F(38), *g3_em1b = F(39), *g3_em1w = F(40),
                *g3_em2b = F(41), *g3_em2w = F(42), *g3_gb = F(43), *g3_gw = F(44),
                *g3_npb = F(45), *g3_npw = F(46);
    const float *gate1_b = F(47), *gate1_w = F(48);
    const float *gate2_b = F(49), *gate2_w = F(50);
    const float *lp_b = F(51), *lp_w = F(52);
    const float *nd1_b = F(53), *nd1_w = F(54);
    const float *nd2_b = F(55), *nd2_w = F(56);
    const float *nep_b = F(57), *nep_w = F(58);
    const float *st1_b = F(59), *st1_w = F(60);
    const float *st2_b = F(61), *st2_w = F(62);
    const int* srcI = (const int*)d_in[63];

    // --- workspace carve-up (all sizes multiples of 4 floats -> 16B aligned) ---
    float* wsp = (float*)d_ws;
    size_t off = 0;
    auto alloc = [&](size_t n) { float* p = wsp + off; off += n; return p; };
    float* hA    = alloc((size_t)NODES * 1024);
    float* hB    = alloc((size_t)NODES * 1024);
    float* hpre  = alloc((size_t)NODES * 1024);
    float* proj  = alloc((size_t)NODES * 512);
    float* temp  = alloc((size_t)NODES * 516);   // max padded ldt (515 -> 516)
    float* ef    = alloc((size_t)EDGES * 3);
    float* deg   = alloc(NODES);
    float* el    = alloc(NODES * 4);
    float* er    = alloc(NODES * 4);
    float* mbuf  = alloc(NODES * 4);
    float* sbuf  = alloc(NODES * 4);
    float* elr   = alloc((size_t)EDGES * 4);
    float* gateh = alloc((size_t)NODES * 128);
    float* gate  = alloc(NODES);
    float* gm    = alloc(BG);
    float* gsum  = alloc(BG);
    float* gemb  = alloc(BG * 1024);
    float* zp    = alloc(BG * 256);
    float* nep   = alloc((size_t)NODES * 256);
    float* ndy   = alloc((size_t)NODES * 128);
    float* zc    = alloc(BG * 128);
    float* ztop  = alloc(BG * 128);
    float* zbot  = alloc(BG * 128);
    float* enh   = alloc(BG * 128);
    float* sth   = alloc(BG * 128);

    // --- output carve-up (tuple flattened in return order) ---
    float* out        = (float*)d_out;
    float* muO        = out;                 // [8,64]
    float* logvarO    = out + 512;           // [8,64]
    float* zO         = out + 1024;          // [8,64]
    float* reconNodeO = out + 1536;          // [10000,4]
    float* reconEdgeO = out + 41536;         // [160000,3]
    float* energyO    = out + 521536;        // [8,2]
    float* stressO    = out + 521552;        // [8,9]

    auto fill = [&](float* p, long n, float v) {
        fill_kernel<<<cdivl(n, 256), 256, 0, stream>>>(p, n, v);
    };
    auto gemm = [&](const float* A, int lda, const float* B, float* C,
                    const float* bias, int M, int K, int Nc, int relu) {
        dim3 g((Nc + WBN - 1) / WBN, (M + WBM - 1) / WBM);
        gemm_wmma_kernel<<<g, 256, 0, stream>>>(A, lda, B, Nc, bias, C, M, K, Nc, relu);
    };
    auto slin = [&](const float* A, const float* W, const float* bias, float* C,
                    int M, int K, int Nc, int relu) {
        small_linear_kernel<<<cdivl((long)M * Nc, 256), 256, 0, stream>>>(A, W, bias, C, M, K, Nc, relu);
    };

    // degree is layer-invariant: compute once
    fill(deg, NODES, 0.0f);
    deg_kernel<<<cdivl(EDGES, 256), 256, 0, stream>>>(srcI, dstI, deg, EDGES);

    auto run_gat = [&](const float* h_in, int fin, int outd,
                       const float* em1w, const float* em1b, const float* em2w, const float* em2b,
                       const float* npw, const float* npb, const float* gw_,
                       const float* al, const float* ar, const float* gb, float* h_out) {
        const int K   = fin + 3;
        const int ldt = (K + 3) & ~3;         // pad to 4 floats for b128 staging
        const int HD  = HH * outd;
        edge_mlp_kernel<<<cdivl(EDGES, 256), 256, 0, stream>>>(edge_feat, em1w, em1b, em2w, em2b, ef, EDGES);
        if (fin >= 32) gemm(h_in, fin, npw, proj, npb, NODES, fin, fin, 0);
        else           slin(h_in, npw, npb, proj, NODES, fin, fin, 0);
        fill(temp, (long)NODES * ldt, 0.0f);
        temp_proj_kernel<<<cdivl((long)NODES * fin, 256), 256, 0, stream>>>(proj, deg, temp, NODES, fin, ldt);
        temp_edge_kernel<<<cdivl(EDGES, 256), 256, 0, stream>>>(ef, srcI, dstI, temp, EDGES, fin, ldt);
        gemm_wmma_kernel<<<dim3((HD + WBN - 1) / WBN, (NODES + WBM - 1) / WBM), 256, 0, stream>>>(
            temp, ldt, gw_, HD, nullptr, hpre, NODES, K, HD, 0);
        attn_lr_kernel<<<cdivl(NODES * 4, 256), 256, 0, stream>>>(hpre, al, ar, el, er, NODES, outd);
        fill(mbuf, NODES * 4, -INFINITY);
        fill(sbuf, NODES * 4, 0.0f);
        edge_logit_max_kernel<<<cdivl(EDGES, 256), 256, 0, stream>>>(srcI, dstI, el, er, elr, mbuf, EDGES);
        edge_exp_sum_kernel<<<cdivl(EDGES, 256), 256, 0, stream>>>(dstI, elr, mbuf, sbuf, EDGES);
        fill(h_out, (long)NODES * HD, 0.0f);
        agg_kernel<<<cdivl((long)EDGES * 4, 256), 256, 0, stream>>>(hpre, elr, sbuf, srcI, dstI, h_out, EDGES, outd);
        elu_bias_kernel<<<cdivl((long)NODES * HD, 256), 256, 0, stream>>>(h_out, gb, (long)NODES * HD, HD);
    };

    // --- encoder: 3 GAT layers ---
    run_gat(node_feat, 4,   64,  g1_em1w, g1_em1b, g1_em2w, g1_em2b, g1_npw, g1_npb, g1_gw, g1_al, g1_ar, g1_gb, hA); // [N,256]
    run_gat(hA,        256, 128, g2_em1w, g2_em1b, g2_em2w, g2_em2b, g2_npw, g2_npb, g2_gw, g2_al, g2_ar, g2_gb, hB); // [N,512]
    run_gat(hB,        512, 256, g3_em1w, g3_em1b, g3_em2w, g3_em2b, g3_npw, g3_npb, g3_gw, g3_al, g3_ar, g3_gb, hA); // [N,1024]

    // --- global attention pooling ---
    gemm(hA, 1024, gate1_w, gateh, gate1_b, NODES, 1024, 128, 1);    // relu fused
    slin(gateh, gate2_w, gate2_b, gate, NODES, 128, 1, 0);
    fill(gm, BG, -INFINITY);
    graph_max_kernel<<<cdivl(NODES, 256), 256, 0, stream>>>(gate, gid, gm, NODES);
    fill(gsum, BG, 0.0f);
    graph_expsum_kernel<<<cdivl(NODES, 256), 256, 0, stream>>>(gate, gid, gm, gsum, NODES);
    fill(gemb, BG * 1024, 0.0f);
    pool_kernel<<<cdivl((long)NODES * 1024, 256), 256, 0, stream>>>(hA, gate, gsum, gid, gemb, NODES, 1024);

    // --- VAE head ---
    slin(gemb, fmu_w, fmu_b, muO,     BG, 1024, LATD, 0);
    slin(gemb, flv_w, flv_b, logvarO, BG, 1024, LATD, 0);
    z_kernel<<<cdivl(BG * LATD, 256), 256, 0, stream>>>(muO, logvarO, eps, zO, BG * LATD);

    // --- decoder ---
    slin(zO, lp_w, lp_b, zp, BG, LATD, 256, 1);                      // zp = relu(z@lp)
    gemm(hA, 1024, nep_w, nep, nep_b, NODES, 1024, 256, 1);          // nep = relu(h@W+b)
    slin(zp, nd1_w, nd1_b, zc, BG, 256, HID, 0);                     // zc = zp@W + b (8 rows)
    gemm(nep, 256, nd1_w, ndy, nullptr, NODES, 256, HID, 0);         // nep@W (WMMA)
    add_relu_gather_kernel<<<cdivl((long)NODES * HID, 256), 256, 0, stream>>>(ndy, zc, gid, NODES, HID);
    slin(ndy, nd2_w, nd2_b, reconNodeO, NODES, HID, 4, 0);
    // edge decoder via 8-row factorization of ed1
    slin(zp, ed1_w,             ed1_b,   ztop, BG, 256, HID, 0);
    slin(zp, ed1_w + 256 * 128, nullptr, zbot, BG, 256, HID, 0);
    edge_decode_kernel<<<cdivl(EDGES, 256), 256, 0, stream>>>(srcI, dstI, gid, ztop, zbot, ed2_w, ed2_b, reconEdgeO, EDGES);
    // energy / stress heads
    slin(zO, en1_w, en1_b, enh, BG, LATD, HID, 1);
    slin(enh, en2_w, en2_b, energyO, BG, HID, 2, 0);
    slin(zO, st1_w, st1_b, sth, BG, LATD, HID, 1);
    slin(sth, st2_w, st2_b, stressO, BG, HID, 9, 0);
}
